// PhysicsInformedSINDyModel_15539191677154
// MI455X (gfx1250) — matrix operand order repaired and checked by reading, not compiled, not run on hardware
//
#include <hip/hip_runtime.h>

// ---------------------------------------------------------------------------
// CDNA5 (gfx1250) fused CNN + SINDy pipeline, wave32 + v_wmma_f32_16x16x32_f16
// ---------------------------------------------------------------------------

typedef __attribute__((ext_vector_type(16))) _Float16 v16h;
typedef __attribute__((ext_vector_type(8)))  float    v8f;
typedef __attribute__((ext_vector_type(4)))  float    v4f;

union HF { v16h h; v4f f4[2]; };   // 32B fragment <-> two 16B loads

__device__ __forceinline__ v8f wmma_f16(v16h a, v16h b, v8f c) {
  // D = A(16x32 f16) x B(32x16 f16) + C(16x16 f32)
  return __builtin_amdgcn_wmma_f32_16x16x32_f16(
      /*neg_a=*/false, a, /*neg_b=*/false, b,
      /*c_mod=*/(short)0, c, /*reuse_a=*/false, /*reuse_b=*/false);
}

// ---------------------------------------------------------------------------
// conv1: (B,1,360) -> relu -> x1 f16 laid out [b][pos(180)][ch(16)]
// K=5 is too small for WMMA; pure VALU, one thread per (b,pos).
// ---------------------------------------------------------------------------
__global__ __launch_bounds__(256)
void conv1_kernel(const float* __restrict__ lidar,
                  const float* __restrict__ w1,   // (16,1,5)
                  const float* __restrict__ b1,   // (16)
                  _Float16* __restrict__ x1, int B) {
  __shared__ float w1s[80];
  __shared__ float b1s[16];
  int tid = threadIdx.x;
  if (tid < 80) w1s[tid] = w1[tid];
  if (tid < 16) b1s[tid] = b1[tid];
  __syncthreads();

  long g = (long)blockIdx.x * 256 + tid;
  long total = (long)B * 180;
  if (g >= total) return;
  int b = (int)(g / 180), p = (int)(g % 180);

  float acc[16];
#pragma unroll
  for (int oc = 0; oc < 16; ++oc) acc[oc] = b1s[oc];
  const float* in = lidar + (long)b * 360;
#pragma unroll
  for (int t = 0; t < 5; ++t) {
    int pos = 2 * p + t - 2;
    if (pos >= 0 && pos < 360) {
      float v = in[pos];
#pragma unroll
      for (int oc = 0; oc < 16; ++oc) acc[oc] += v * w1s[oc * 5 + t];
    }
  }
  union { _Float16 h[16]; v4f f4[2]; } o;
#pragma unroll
  for (int oc = 0; oc < 16; ++oc) o.h[oc] = (_Float16)fmaxf(acc[oc], 0.f);
  v4f* dst = (v4f*)(x1 + g * 16);          // 32B aligned
  dst[0] = o.f4[0];
  dst[1] = o.f4[1];
}

// ---------------------------------------------------------------------------
// pack fc weights once: fcw16[oc][k'] f16 with k' = pos*64 + ch (x3's memory
// order), so the fc GEMM reads contiguous b128 B-fragments from global/L2.
// ---------------------------------------------------------------------------
__global__ __launch_bounds__(256)
void pack_fcw_kernel(const float* __restrict__ fc_w,   // (64,2880), k = ch*45+pos
                     _Float16* __restrict__ fcw16) {
  int idx = blockIdx.x * 256 + threadIdx.x;            // 64*2880 = 184320
  if (idx >= 64 * 2880) return;
  int oc = idx / 2880, kp = idx % 2880;
  int ch = kp & 63, pos = kp >> 6;
  fcw16[idx] = (_Float16)fc_w[oc * 2880 + ch * 45 + pos];
}

// ---------------------------------------------------------------------------
// conv2: im2col GEMM, M=B*90 tokens, N=32, K=80 (pad->96 = 3 K-chunks of 32).
// x1 [b][180][16] f16 -> x2 [b][90][32] f16.  One M-tile(16 tokens) per wave.
// Epilogue transposes C through per-wave LDS so stores are b128-coalesced.
// ---------------------------------------------------------------------------
__global__ __launch_bounds__(256)
void conv2_kernel(const _Float16* __restrict__ x1,
                  const float* __restrict__ w2,   // (32,16,5)
                  const float* __restrict__ b2,   // (32)
                  _Float16* __restrict__ x2, int B) {
  __shared__ __align__(16) _Float16 wT[32][96];       // [oc][k = tap*16+ic], pad k>=80
  __shared__ __align__(16) _Float16 stage[8][16][32]; // per-wave C transpose tile
  int tid = threadIdx.x;
  for (int i = tid; i < 32 * 96; i += 256) {
    int oc = i / 96, k = i % 96;
    float v = 0.f;
    if (k < 80) { int tap = k >> 4, ic = k & 15; v = w2[(oc * 16 + ic) * 5 + tap]; }
    wT[oc][k] = (_Float16)v;
  }
  __syncthreads();

  int lane = tid & 31, wave = tid >> 5;
  int hi = lane >> 4;          // K-half select (ISA A/B fragment layout)
  int col = lane & 15;         // M row (for A) / N col (for B,C)
  long tokens = (long)B * 90;
  long tok0 = ((long)blockIdx.x * 8 + wave) * 16;
  if (tok0 >= tokens) return;

  // B fragments: 2 N-tiles x 3 K-chunks, from LDS (two ds_load_b128 each)
  v16h bfr[2][3];
#pragma unroll
  for (int nt = 0; nt < 2; ++nt)
#pragma unroll
    for (int c = 0; c < 3; ++c) {
      HF u;
      const v4f* src = (const v4f*)&wT[nt * 16 + col][c * 32 + hi * 8];
      u.f4[0] = src[0]; u.f4[1] = src[2];
      bfr[nt][c] = u.h;
    }

  // A fragments: this lane's token row = tok0 + col
  long tok = tok0 + col;
  int b = (int)(tok / 90), p = (int)(tok % 90);
  int ch0 = hi * 8;
  v4f z = {0.f, 0.f, 0.f, 0.f};
  v16h afr[3];
#pragma unroll
  for (int c = 0; c < 3; ++c) {
    HF u;
    int t0 = 2 * c, t1 = 2 * c + 1;
    int pos0 = 2 * p + t0 - 2;
    int pos1 = 2 * p + t1 - 2;
    u.f4[0] = (pos0 >= 0 && pos0 < 180)
                  ? *(const v4f*)(x1 + ((long)b * 180 + pos0) * 16 + ch0) : z;
    u.f4[1] = (t1 < 5 && pos1 >= 0 && pos1 < 180)
                  ? *(const v4f*)(x1 + ((long)b * 180 + pos1) * 16 + ch0) : z;
    afr[c] = u.h;
  }

#pragma unroll
  for (int nt = 0; nt < 2; ++nt) {
    v8f acc = {};
#pragma unroll
    for (int c = 0; c < 3; ++c) acc = wmma_f16(afr[c], bfr[nt][c], acc);
    int oc = nt * 16 + col;
    float bias = b2[oc];
#pragma unroll
    for (int r = 0; r < 8; ++r)
      stage[wave][r + hi * 8][oc] = (_Float16)fmaxf(acc[r] + bias, 0.f);
  }
  // wave-local transpose readout: lane -> (token m = col, oc half = hi)
  {
    const v4f* src = (const v4f*)&stage[wave][col][hi * 16];
    v4f d0 = src[0], d1 = src[1];            // ds_load_b128 x2 (in-order LDS pipe)
    v4f* dst = (v4f*)(x2 + (tok0 + col) * 32 + hi * 16);
    dst[0] = d0; dst[1] = d1;                // global_store_b128 x2
  }
}

// ---------------------------------------------------------------------------
// conv3: M=B*45, N=64 (4 N-tiles), K=160 = exactly 5 K-chunks (1 tap each).
// x2 [b][90][32] -> x3 [b][45][64] f16 (k' = pos*64+ch ordering for fc).
// ---------------------------------------------------------------------------
__global__ __launch_bounds__(256)
void conv3_kernel(const _Float16* __restrict__ x2,
                  const float* __restrict__ w3,   // (64,32,5)
                  const float* __restrict__ b3,   // (64)
                  _Float16* __restrict__ x3, int B) {
  __shared__ __align__(16) _Float16 wT[64][160];      // [oc][k = tap*32+ic]
  __shared__ __align__(16) _Float16 stage[8][16][64]; // per-wave C transpose tile
  int tid = threadIdx.x;
  for (int i = tid; i < 64 * 160; i += 256) {
    int oc = i / 160, k = i % 160;
    int tap = k >> 5, ic = k & 31;
    wT[oc][k] = (_Float16)w3[(oc * 32 + ic) * 5 + tap];
  }
  __syncthreads();

  int lane = tid & 31, wave = tid >> 5, hi = lane >> 4, col = lane & 15;
  long tokens = (long)B * 45;
  long tok0 = ((long)blockIdx.x * 8 + wave) * 16;
  if (tok0 >= tokens) return;

  long tok = tok0 + col;
  int b = (int)(tok / 45), p = (int)(tok % 45);
  int ch0 = hi * 8;
  v4f z = {0.f, 0.f, 0.f, 0.f};
  v16h afr[5];
#pragma unroll
  for (int t = 0; t < 5; ++t) {
    HF u;
    int pos = 2 * p + t - 2;
    if (pos >= 0 && pos < 90) {
      const _Float16* src = x2 + ((long)b * 90 + pos) * 32;
      u.f4[0] = *(const v4f*)(src + ch0);
      u.f4[1] = *(const v4f*)(src + ch0 + 16);
    } else { u.f4[0] = z; u.f4[1] = z; }
    afr[t] = u.h;
  }

#pragma unroll
  for (int nt = 0; nt < 4; ++nt) {
    v8f acc = {};
#pragma unroll
    for (int t = 0; t < 5; ++t) {
      HF u;
      const v4f* src = (const v4f*)&wT[nt * 16 + col][t * 32 + hi * 8];
      u.f4[0] = src[0]; u.f4[1] = src[2];
      acc = wmma_f16(afr[t], u.h, acc);
    }
    int oc = nt * 16 + col;
    float bias = b3[oc];
#pragma unroll
    for (int r = 0; r < 8; ++r)
      stage[wave][r + hi * 8][oc] = (_Float16)fmaxf(acc[r] + bias, 0.f);
  }
  // wave-local transpose readout: lane -> (token m = col, oc half = hi*32)
  {
    const v4f* src = (const v4f*)&stage[wave][col][hi * 32];
    v4f d0 = src[0], d1 = src[1], d2 = src[2], d3 = src[3];
    v4f* dst = (v4f*)(x3 + (tok0 + col) * 64 + hi * 32);
    dst[0] = d0; dst[1] = d1; dst[2] = d2; dst[3] = d3;
  }
}

// ---------------------------------------------------------------------------
// fc + SINDy (fused): 32 samples per block.  GEMM M=16x2, N=64, K=2880
// (90 chunks of 32).  A rows stream from x3; B rows stream from the packed
// f16 weight image (360 KB, L2-resident) -- no barriers in the K loop.
// Then library = [1,state,quad(triu),sin,cos], dxyz = library @ sindy_w.T,
// out = odom + 0.1*dxyz.  Reduction via LDS ds_add_f32 atomics.
// ---------------------------------------------------------------------------
__global__ __launch_bounds__(256)
void fc_sindy_kernel(const _Float16* __restrict__ x3,
                     const _Float16* __restrict__ fcw16, // (64,2880) packed, k' order
                     const float* __restrict__ fc_b,     // (64)
                     const float* __restrict__ odom,     // (B,3)
                     const float* __restrict__ sw,       // (3,2480)
                     float* __restrict__ out, int B) {
  __shared__ float state[32][72];                      // z(64) + odom(3), padded
  __shared__ float red[32][4];

  int tid = threadIdx.x;
  int lane = tid & 31, wave = tid >> 5, hi = lane >> 4, col = lane & 15;
  int mt = wave >> 2;            // 0..1 : M-tile of 16 samples
  int nt = wave & 3;             // 0..3 : N-tile of 16 outputs
  long s0 = (long)blockIdx.x * 32;
  const _Float16* arow = x3 + (s0 + mt * 16 + col) * 2880;      // A: sample row
  const _Float16* brow = fcw16 + (long)(nt * 16 + col) * 2880;  // B: oc row

  v8f acc = {};
  for (int c = 0; c < 90; ++c) {
    HF a, bf;
    int aoff = c * 32 + hi * 8;        // A: lanes<16 K 0-7/16-23, >=16 K 8-15/24-31
    a.f4[0] = *(const v4f*)(arow + aoff);
    a.f4[1] = *(const v4f*)(arow + aoff + 16);
    int boff = c * 32 + hi * 8;        // B: lanes<16 K 0-7/16-23, >=16 K 8-15/24-31
    bf.f4[0] = *(const v4f*)(brow + boff);
    bf.f4[1] = *(const v4f*)(brow + boff + 16);
    if (c + 1 < 90) {
      __builtin_prefetch(arow + (c + 1) * 32, 0, 1);
      __builtin_prefetch(brow + (c + 1) * 32, 0, 1);
    }
    acc = wmma_f16(a.h, bf.h, acc);
  }

  // scatter z (+bias) into state; append odom; zero reductions
  {
    int n = nt * 16 + col;
    float bias = fc_b[n];
#pragma unroll
    for (int r = 0; r < 8; ++r) {
      int m = mt * 16 + r + hi * 8;
      state[m][n] = acc[r] + bias;
    }
  }
  if (tid < 96) {
    int s = tid / 3, o = tid % 3;
    state[s][64 + o] = odom[(s0 + s) * 3 + o];
    red[s][o] = 0.f;
  }
  __syncthreads();

  // SINDy: each wave handles 4 samples; lanes split library terms
  for (int s = wave; s < 32; s += 8) {
    const float* st = state[s];
    float a0 = 0.f, a1 = 0.f, a2 = 0.f;
    // quadratic (upper-triangular) terms; library idx = 68 + base(r) + (c-r)
    for (int r = lane; r < 67; r += 32) {
      int base = 68 + r * 67 - (r * (r - 1)) / 2;
      float sr = st[r];
      for (int cc = r; cc < 67; ++cc) {
        float q = sr * st[cc];
        int idx = base + (cc - r);
        a0 += sw[idx] * q;
        a1 += sw[2480 + idx] * q;
        a2 += sw[4960 + idx] * q;
      }
    }
    // linear + sin + cos terms
    for (int i = lane; i < 67; i += 32) {
      float si = st[i];
      float sn = __sinf(si), cs = __cosf(si);
      int il = 1 + i;
      int is = 2346 + i;   // 1+67+2278
      int ic = 2413 + i;   // +67
      a0 += sw[il] * si + sw[is] * sn + sw[ic] * cs;
      a1 += sw[2480 + il] * si + sw[2480 + is] * sn + sw[2480 + ic] * cs;
      a2 += sw[4960 + il] * si + sw[4960 + is] * sn + sw[4960 + ic] * cs;
    }
    if (lane == 0) { a0 += sw[0]; a1 += sw[2480]; a2 += sw[4960]; }
    atomicAdd(&red[s][0], a0);   // ds_add_f32
    atomicAdd(&red[s][1], a1);
    atomicAdd(&red[s][2], a2);
  }
  __syncthreads();

  if (tid < 96) {
    int s = tid / 3, o = tid % 3;
    long gs = s0 + s;
    out[gs * 3 + o] = odom[gs * 3 + o] + 0.1f * red[s][o];
  }
}

// ---------------------------------------------------------------------------
extern "C" void kernel_launch(void* const* d_in, const int* in_sizes, int n_in,
                              void* d_out, int out_size, void* d_ws, size_t ws_size,
                              hipStream_t stream) {
  (void)n_in; (void)out_size; (void)ws_size;
  const float* lidar = (const float*)d_in[0];
  const float* odom  = (const float*)d_in[1];
  const float* w1    = (const float*)d_in[2];
  const float* b1    = (const float*)d_in[3];
  const float* w2    = (const float*)d_in[4];
  const float* b2    = (const float*)d_in[5];
  const float* w3    = (const float*)d_in[6];
  const float* b3    = (const float*)d_in[7];
  const float* fcw   = (const float*)d_in[8];
  const float* fcb   = (const float*)d_in[9];
  const float* sw    = (const float*)d_in[10];
  float* out = (float*)d_out;
  int B = in_sizes[0] / 360;

  // x1 / x2 / x3 are all B*2880 f16 elements; ping-pong two ws buffers,
  // plus a 360KB packed fc weight image.
  size_t S = (size_t)B * 2880 * sizeof(_Float16);
  _Float16* x1    = (_Float16*)d_ws;
  _Float16* x2    = (_Float16*)((char*)d_ws + S);
  _Float16* fcw16 = (_Float16*)((char*)d_ws + 2 * S);
  _Float16* x3    = x1;   // x1 is dead once conv2 completes

  pack_fcw_kernel<<<(64 * 2880 + 255) / 256, 256, 0, stream>>>(fcw, fcw16);
  {
    long total = (long)B * 180;
    int blocks = (int)((total + 255) / 256);
    conv1_kernel<<<blocks, 256, 0, stream>>>(lidar, w1, b1, x1, B);
  }
  {
    long tiles = ((long)B * 90 + 15) / 16;
    int blocks = (int)((tiles + 7) / 8);
    conv2_kernel<<<blocks, 256, 0, stream>>>(x1, w2, b2, x2, B);
  }
  {
    long tiles = ((long)B * 45 + 15) / 16;
    int blocks = (int)((tiles + 7) / 8);
    conv3_kernel<<<blocks, 256, 0, stream>>>(x2, w3, b3, x3, B);
  }
  {
    int blocks = (B + 31) / 32;
    fc_sindy_kernel<<<blocks, 256, 0, stream>>>(x3, fcw16, fcb, odom, sw, out, B);
  }
}
